// Critic_7576322310714
// MI455X (gfx1250) — matrix operand order repaired and checked by reading, not compiled, not run on hardware
//
#include <hip/hip_runtime.h>
#include <hip/hip_bf16.h>

#define N_NODES  4000
#define N_EDGES  12000
#define N_GRAPHS 128
#define NODE_F   16
#define EDGE_F   4

typedef unsigned short u16;
typedef __attribute__((ext_vector_type(16))) __bf16 v16bf;
typedef __attribute__((ext_vector_type(8)))  float  v8f;

union FragBF { uint4 q[2]; v16bf v; };

__device__ __forceinline__ u16 f2bf(float f) {
    unsigned u; __builtin_memcpy(&u, &f, 4);
    u += 0x7FFFu + ((u >> 16) & 1u);
    return (u16)(u >> 16);
}
__device__ __forceinline__ float lrelu(float v) { return v >= 0.f ? v : 0.01f * v; }

// packed bf16 multiply with bf16 RNE round (VOP3P V_PK_MUL_BF16)
__device__ __forceinline__ unsigned pk_mul_bf16(unsigned a, unsigned b) {
    unsigned d;
    asm("v_pk_mul_bf16 %0, %1, %2" : "=v"(d) : "v"(a), "v"(b));
    return d;
}

// ---------------------------------------------------------------- elementwise
__global__ void k_zero(float* p, int n) {
    int i = blockIdx.x * blockDim.x + threadIdx.x;
    if (i < n) p[i] = 0.f;
}

__global__ void k_f32_to_bf16(const float* __restrict__ in, u16* __restrict__ out, int n) {
    int i = blockIdx.x * blockDim.x + threadIdx.x;
    if (i < n) out[i] = f2bf(in[i]);
}

__global__ void k_count_edges(const int* __restrict__ dst, float* __restrict__ cnt) {
    int e = blockIdx.x * blockDim.x + threadIdx.x;
    if (e < N_EDGES) atomicAdd(&cnt[dst[e]], 1.f);
}

__global__ void k_count_graphs(const int* __restrict__ batch, float* __restrict__ gcnt) {
    int n = blockIdx.x * blockDim.x + threadIdx.x;
    if (n < N_NODES) atomicAdd(&gcnt[batch[n]], 1.f);
}

// edge MLP first layer: eh[e,j] = lrelu(ea[e]@w1 + b1), plus appended constant-1 column
template<int H>
__global__ void k_edge_hidden(const float* __restrict__ ea, const float* __restrict__ w1,
                              const float* __restrict__ b1, u16* __restrict__ ehx) {
    int idx = blockIdx.x * blockDim.x + threadIdx.x;
    if (idx >= N_EDGES * (H + 1)) return;
    int e = idx / (H + 1), j = idx - e * (H + 1);
    float v;
    if (j < H) {
        v = b1[j];
        #pragma unroll
        for (int q = 0; q < EDGE_F; ++q) v += ea[e * EDGE_F + q] * w1[q * H + j];
        v = lrelu(v);
    } else {
        v = 1.f;   // folds b2 rows into the GEMM
    }
    ehx[idx] = f2bf(v);
}

// Pack w2 [H, IN*OUT] (+ b2 as extra rows, + zero pad to K%32==0) into bf16 WMMA-B
// fragment layout: frag(c,t): 32 lanes x 16 halves; lane = (klocal/16)*16 + n%16, elem = klocal%16.
template<int IN, int HID, int OUT>
__global__ void k_pack_w2(const float* __restrict__ w2, const float* __restrict__ b2,
                          u16* __restrict__ w2x) {
    constexpr int KR  = (HID + 1) * IN;
    constexpr int NC  = (KR + 31) / 32;
    constexpr int K32 = NC * 32;
    constexpr int NT  = OUT / 16;
    int idx = blockIdx.x * blockDim.x + threadIdx.x;
    if (idx >= K32 * OUT) return;
    int kp = idx / OUT, o = idx - kp * OUT;
    float v = 0.f;
    if (kp < HID * IN) {
        int h = kp / IN, i = kp - h * IN;
        v = w2[(size_t)h * (IN * OUT) + i * OUT + o];
    } else if (kp < KR) {
        int i = kp - HID * IN;
        v = b2[i * OUT + o];
    }
    int c = kp >> 5, kl = kp & 31;
    int lane = ((kl >> 4) << 4) | (o & 15);
    int e    = kl & 15;
    int t    = o >> 4;
    size_t off = (((size_t)(c * NT + t)) * 32 + lane) * 16 + e;
    w2x[off] = f2bf(v);
}

// ---------------------------------------------------------------- fused NNConv GEMM
// msg[e,o] = sum_k eh'[e,k/IN] * hsrc[e,k%IN] * W2x[k,o]   (bias rows folded into K)
// A tile (32 edges x 32 K) built cooperatively in LDS each chunk; bf16 WMMA, f32 acc.
// Since IN % 8 == 0 and each thread's 8-wide K group starts at a multiple of 8, a group
// never crosses an h boundary: 1 splatted-eh b32 + 1 hs b128 + 4 v_pk_mul_bf16, no guards
// (K-pad region multiplies a zeroed eh pad column; B is zero there anyway).
template<int IN, int HID, int OUT, int KSPLIT>
__global__ __launch_bounds__(128) void k_nnconv_gemm(
        const u16* __restrict__ ehx, const u16* __restrict__ hbf,
        const int* __restrict__ srcIdx, const u16* __restrict__ w2x,
        float* __restrict__ msgP) {
    constexpr int KR  = (HID + 1) * IN;
    constexpr int NC  = (KR + 31) / 32;
    constexpr int NT  = OUT / 16;
    constexpr int NPW = NT / 4;                 // n-tiles per wave (4 waves)
    constexpr int CPS = (NC + KSPLIT - 1) / KSPLIT;
    constexpr int EHS = HID + 2;                // splatted eh row stride (uints)
    constexpr int HSS = IN + 8;                 // hs row stride (ushorts, multiple of 8)

    __shared__ unsigned sEhS[32 * EHS];          // eh value splatted into both bf16 halves
    __shared__ __align__(16) u16 sHs[32 * HSS];
    __shared__ __align__(16) u16 sA[2 * 32 * 40]; // double-buffered 32x32 bf16 A tile
    __shared__ int sSrc[32];

    const int tid  = threadIdx.x;
    const int lane = tid & 31;
    const int w    = tid >> 5;
    const int e0   = blockIdx.x * 32;

    if (tid < 32) sSrc[tid] = srcIdx[e0 + tid];
    __syncthreads();
    for (int idx = tid; idx < 32 * (HID + 1); idx += 128) {
        int r = idx / (HID + 1), j = idx - r * (HID + 1);
        unsigned hv = ehx[(size_t)(e0 + r) * (HID + 1) + j];
        sEhS[r * EHS + j] = hv | (hv << 16);
    }
    if (tid < 32) sEhS[tid * EHS + (HID + 1)] = 0;   // K-pad h column -> 0
    for (int idx = tid; idx < 32 * IN; idx += 128) {
        int r = idx / IN, i = idx - r * IN;
        sHs[r * HSS + i] = hbf[(size_t)sSrc[r] * IN + i];
    }
    __syncthreads();

    v8f zero8 = {0.f, 0.f, 0.f, 0.f, 0.f, 0.f, 0.f, 0.f};
    v8f acc[2][NPW];
    #pragma unroll
    for (int a = 0; a < 2; ++a)
        #pragma unroll
        for (int b = 0; b < NPW; ++b) acc[a][b] = zero8;

    const int am      = tid >> 2;          // A-build: edge row 0..31
    const int akq     = (tid & 3) << 3;    // A-build: K sub-group of 8
    const int laneOff = (lane >> 4) << 4;  // A-frag K offset per lane half
    const int mrow    = lane & 15;

    const int cBeg = blockIdx.y * CPS;
    const int cEnd = (cBeg + CPS < NC) ? (cBeg + CPS) : NC;

    for (int c = cBeg; c < cEnd; ++c) {
        u16* sAbuf = sA + (c & 1) * (32 * 40);
        {   // guard-free A build: group of 8 K stays within one h
            int kb = c * 32 + akq;
            int h  = kb / IN;
            int i  = kb - h * IN;                       // multiple of 8
            unsigned ehp = sEhS[am * EHS + h];
            uint4 hsv = *(const uint4*)&sHs[am * HSS + i];
            uint4 r;
            r.x = pk_mul_bf16(hsv.x, ehp);
            r.y = pk_mul_bf16(hsv.y, ehp);
            r.z = pk_mul_bf16(hsv.z, ehp);
            r.w = pk_mul_bf16(hsv.w, ehp);
            *(uint4*)&sAbuf[am * 40 + akq] = r;
        }
        __syncthreads();   // single barrier per chunk (sA double-buffered)

        // A fragments (contiguous K to match packed B: lanes<16 hold K 0-15, lanes>=16 hold 16-31)
        FragBF a0, a1;
        a0.q[0] = *(const uint4*)&sAbuf[mrow * 40 + laneOff];
        a0.q[1] = *(const uint4*)&sAbuf[mrow * 40 + laneOff + 8];
        a1.q[0] = *(const uint4*)&sAbuf[(16 + mrow) * 40 + laneOff];
        a1.q[1] = *(const uint4*)&sAbuf[(16 + mrow) * 40 + laneOff + 8];

        #pragma unroll
        for (int j = 0; j < NPW; ++j) {
            int t = w * NPW + j;
            const uint4* bp = (const uint4*)w2x + ((size_t)(c * NT + t) * 64) + lane * 2;
            FragBF b; b.q[0] = bp[0]; b.q[1] = bp[1];
            acc[0][j] = __builtin_amdgcn_wmma_f32_16x16x32_bf16(
                false, a0.v, false, b.v, (short)0, acc[0][j], false, false);
            acc[1][j] = __builtin_amdgcn_wmma_f32_16x16x32_bf16(
                false, a1.v, false, b.v, (short)0, acc[1][j], false, false);
        }
    }

    // store K-split partials (C layout: elem r -> row r + 8*(lane/16), col lane%16)
    const int p = blockIdx.y;
    #pragma unroll
    for (int mf = 0; mf < 2; ++mf)
        #pragma unroll
        for (int j = 0; j < NPW; ++j) {
            int t = w * NPW + j;
            int o = t * 16 + (lane & 15);
            #pragma unroll
            for (int r = 0; r < 8; ++r) {
                int mm = mf * 16 + r + ((lane >> 4) << 3);
                msgP[((size_t)p * N_EDGES + (e0 + mm)) * OUT + o] = acc[mf][j][r];
            }
        }
}

// reduce K-split partials + scatter-add to destination nodes
template<int OUT, int P>
__global__ void k_combine_scatter(const float* __restrict__ msgP, const int* __restrict__ dstIdx,
                                  float* __restrict__ agg) {
    int idx = blockIdx.x * blockDim.x + threadIdx.x;
    if (idx >= N_EDGES * OUT) return;
    int e = idx / OUT, o = idx - e * OUT;
    float s = 0.f;
    #pragma unroll
    for (int p = 0; p < P; ++p) s += msgP[((size_t)p * N_EDGES + e) * OUT + o];
    atomicAdd(&agg[(size_t)dstIdx[e] * OUT + o], s);
}

// scatter-mean + root term + bias + lrelu, then concat x; write f32 and bf16 copies
template<int INDIM, int OUT>
__global__ void k_node_update(const float* __restrict__ agg, const float* __restrict__ cnt,
                              const float* __restrict__ hin, const float* __restrict__ root,
                              const float* __restrict__ bias, const float* __restrict__ x,
                              float* __restrict__ hcat, u16* __restrict__ hcatb) {
    constexpr int CAT = OUT + NODE_F;
    int idx = blockIdx.x * blockDim.x + threadIdx.x;
    if (idx >= N_NODES * CAT) return;
    int n = idx / CAT, o = idx - n * CAT;
    float v;
    if (o < OUT) {
        float s = agg[(size_t)n * OUT + o] / fmaxf(cnt[n], 1.f) + bias[o];
        for (int i = 0; i < INDIM; ++i) s += hin[(size_t)n * INDIM + i] * root[i * OUT + o];
        v = lrelu(s);
    } else {
        v = x[n * NODE_F + (o - OUT)];
    }
    hcat[idx]  = v;
    hcatb[idx] = f2bf(v);
}

__global__ void k_pool(const float* __restrict__ h3, const int* __restrict__ batch,
                       float* __restrict__ pooled) {
    int idx = blockIdx.x * blockDim.x + threadIdx.x;
    if (idx >= N_NODES * 272) return;
    int n = idx / 272, o = idx - n * 272;
    atomicAdd(&pooled[batch[n] * 272 + o], h3[idx]);
}

__global__ void k_pool_div(float* __restrict__ pooled, const float* __restrict__ gcnt) {
    int idx = blockIdx.x * blockDim.x + threadIdx.x;
    if (idx >= N_GRAPHS * 272) return;
    pooled[idx] /= fmaxf(gcnt[idx / 272], 1.f);
}

template<int INF, int OUTF, bool ACT>
__global__ void k_linear(const float* __restrict__ in, const float* __restrict__ wgt,
                         const float* __restrict__ b, float* __restrict__ out, int rows) {
    int idx = blockIdx.x * blockDim.x + threadIdx.x;
    if (idx >= rows * OUTF) return;
    int r = idx / OUTF, o = idx - r * OUTF;
    float s = b[o];
    for (int i = 0; i < INF; ++i) s += in[(size_t)r * INF + i] * wgt[i * OUTF + o];
    out[idx] = ACT ? lrelu(s) : s;
}

#define CDIV(a, b) (((a) + (b) - 1) / (b))

extern "C" void kernel_launch(void* const* d_in, const int* in_sizes, int n_in,
                              void* d_out, int out_size, void* d_ws, size_t ws_size,
                              hipStream_t stream) {
    (void)in_sizes; (void)n_in; (void)out_size; (void)ws_size;
    const float* x    = (const float*)d_in[0];
    const int*   ei   = (const int*)  d_in[1];
    const float* ea   = (const float*)d_in[2];
    const int*   bat  = (const int*)  d_in[3];
    const float* m1w1 = (const float*)d_in[4];  const float* m1b1 = (const float*)d_in[5];
    const float* m1w2 = (const float*)d_in[6];  const float* m1b2 = (const float*)d_in[7];
    const float* root1= (const float*)d_in[8];  const float* bias1= (const float*)d_in[9];
    const float* m2w1 = (const float*)d_in[10]; const float* m2b1 = (const float*)d_in[11];
    const float* m2w2 = (const float*)d_in[12]; const float* m2b2 = (const float*)d_in[13];
    const float* root2= (const float*)d_in[14]; const float* bias2= (const float*)d_in[15];
    const float* m3w1 = (const float*)d_in[16]; const float* m3b1 = (const float*)d_in[17];
    const float* m3w2 = (const float*)d_in[18]; const float* m3b2 = (const float*)d_in[19];
    const float* root3= (const float*)d_in[20]; const float* bias3= (const float*)d_in[21];
    const float* fc1w = (const float*)d_in[22]; const float* fc1b = (const float*)d_in[23];
    const float* fc2w = (const float*)d_in[24]; const float* fc2b = (const float*)d_in[25];
    const float* fc3w = (const float*)d_in[26]; const float* fc3b = (const float*)d_in[27];

    const int* src = ei;
    const int* dst = ei + N_EDGES;

    char* base = (char*)d_ws;
    size_t off = 0;
    auto alloc = [&](size_t bytes) -> void* {
        void* p = base + off;
        off = (off + bytes + 255) & ~(size_t)255;
        return p;
    };

    // K sizes: L1: (128+1)*16=2064 -> 65 chunks (2080); L2/L3: (256+1)*144=37008 -> 1157 chunks (37024)
    u16* ehx1 = (u16*)alloc((size_t)N_EDGES * 129 * 2);
    u16* ehx2 = (u16*)alloc((size_t)N_EDGES * 257 * 2);
    u16* ehx3 = (u16*)alloc((size_t)N_EDGES * 257 * 2);
    u16* w2x1 = (u16*)alloc((size_t)2080  * 128 * 2);
    u16* w2x2 = (u16*)alloc((size_t)37024 * 128 * 2);
    u16* w2x3 = (u16*)alloc((size_t)37024 * 256 * 2);
    u16* xbf  = (u16*)alloc((size_t)N_NODES * 16 * 2);
    float* h1   = (float*)alloc((size_t)N_NODES * 144 * 4);
    u16*   h1b  = (u16*)  alloc((size_t)N_NODES * 144 * 2);
    float* h2   = (float*)alloc((size_t)N_NODES * 144 * 4);
    u16*   h2b  = (u16*)  alloc((size_t)N_NODES * 144 * 2);
    float* h3   = (float*)alloc((size_t)N_NODES * 272 * 4);
    u16*   h3b  = (u16*)  alloc((size_t)N_NODES * 272 * 2);
    float* msgP = (float*)alloc((size_t)4 * N_EDGES * 256 * 4);
    float* agg  = (float*)alloc((size_t)N_NODES * 256 * 4);
    float* cnt  = (float*)alloc((size_t)N_NODES * 4);
    float* gcnt = (float*)alloc((size_t)N_GRAPHS * 4);
    float* pooled = (float*)alloc((size_t)N_GRAPHS * 272 * 4);
    float* z1   = (float*)alloc((size_t)N_GRAPHS * 256 * 4);
    float* z2   = (float*)alloc((size_t)N_GRAPHS * 512 * 4);

    const int T = 256;

    // counts (shared by all layers) + x in bf16
    k_zero<<<CDIV(N_NODES, T), T, 0, stream>>>(cnt, N_NODES);
    k_zero<<<1, 128, 0, stream>>>(gcnt, N_GRAPHS);
    k_zero<<<CDIV(N_GRAPHS * 272, T), T, 0, stream>>>(pooled, N_GRAPHS * 272);
    k_count_edges<<<CDIV(N_EDGES, T), T, 0, stream>>>(dst, cnt);
    k_count_graphs<<<CDIV(N_NODES, T), T, 0, stream>>>(bat, gcnt);
    k_f32_to_bf16<<<CDIV(N_NODES * 16, T), T, 0, stream>>>(x, xbf, N_NODES * 16);

    // ---- layer 1: IN=16, HID=128, OUT=128
    k_edge_hidden<128><<<CDIV(N_EDGES * 129, T), T, 0, stream>>>(ea, m1w1, m1b1, ehx1);
    k_pack_w2<16, 128, 128><<<CDIV(2080 * 128, T), T, 0, stream>>>(m1w2, m1b2, w2x1);
    k_zero<<<CDIV(N_NODES * 128, T), T, 0, stream>>>(agg, N_NODES * 128);
    k_nnconv_gemm<16, 128, 128, 2><<<dim3(N_EDGES / 32, 2), 128, 0, stream>>>(ehx1, xbf, src, w2x1, msgP);
    k_combine_scatter<128, 2><<<CDIV(N_EDGES * 128, T), T, 0, stream>>>(msgP, dst, agg);
    k_node_update<16, 128><<<CDIV(N_NODES * 144, T), T, 0, stream>>>(agg, cnt, x, root1, bias1, x, h1, h1b);

    // ---- layer 2: IN=144, HID=256, OUT=128
    k_edge_hidden<256><<<CDIV(N_EDGES * 257, T), T, 0, stream>>>(ea, m2w1, m2b1, ehx2);
    k_pack_w2<144, 256, 128><<<CDIV(37024 * 128, T), T, 0, stream>>>(m2w2, m2b2, w2x2);
    k_zero<<<CDIV(N_NODES * 128, T), T, 0, stream>>>(agg, N_NODES * 128);
    k_nnconv_gemm<144, 256, 128, 4><<<dim3(N_EDGES / 32, 4), 128, 0, stream>>>(ehx2, h1b, src, w2x2, msgP);
    k_combine_scatter<128, 4><<<CDIV(N_EDGES * 128, T), T, 0, stream>>>(msgP, dst, agg);
    k_node_update<144, 128><<<CDIV(N_NODES * 144, T), T, 0, stream>>>(agg, cnt, h1, root2, bias2, x, h2, h2b);

    // ---- layer 3: IN=144, HID=256, OUT=256
    k_edge_hidden<256><<<CDIV(N_EDGES * 257, T), T, 0, stream>>>(ea, m3w1, m3b1, ehx3);
    k_pack_w2<144, 256, 256><<<CDIV(37024 * 256, T), T, 0, stream>>>(m3w2, m3b2, w2x3);
    k_zero<<<CDIV(N_NODES * 256, T), T, 0, stream>>>(agg, N_NODES * 256);
    k_nnconv_gemm<144, 256, 256, 4><<<dim3(N_EDGES / 32, 4), 128, 0, stream>>>(ehx3, h2b, src, w2x3, msgP);
    k_combine_scatter<256, 4><<<CDIV(N_EDGES * 256, T), T, 0, stream>>>(msgP, dst, agg);
    k_node_update<144, 256><<<CDIV(N_NODES * 272, T), T, 0, stream>>>(agg, cnt, h2, root3, bias3, x, h3, h3b);

    // ---- pooling + FC head
    k_pool<<<CDIV(N_NODES * 272, T), T, 0, stream>>>(h3, bat, pooled);
    k_pool_div<<<CDIV(N_GRAPHS * 272, T), T, 0, stream>>>(pooled, gcnt);
    k_linear<272, 256, true><<<CDIV(N_GRAPHS * 256, T), T, 0, stream>>>(pooled, fc1w, fc1b, z1, N_GRAPHS);
    k_linear<256, 512, true><<<CDIV(N_GRAPHS * 512, T), T, 0, stream>>>(z1, fc2w, fc2b, z2, N_GRAPHS);
    k_linear<512, 1, false><<<CDIV(N_GRAPHS * 1, T), T, 0, stream>>>(z2, fc3w, fc3b, (float*)d_out, N_GRAPHS);
}
